// KimiDeltaAttention_6640019440434
// MI455X (gfx1250) — compile-verified, matrix operand order checked by reading, not run on hardware
//
#include <hip/hip_runtime.h>
#include <hip/hip_bf16.h>
#include <cstdint>
#include <cstddef>

// ---------------------------------------------------------------------------
// CDNA5 (gfx1250) WMMA helpers: v_wmma_f32_16x16x32_bf16, wave32.
// ---------------------------------------------------------------------------
typedef __attribute__((ext_vector_type(16))) __bf16 v16bf;
typedef __attribute__((ext_vector_type(8)))  __bf16 bf16x8;
typedef __attribute__((ext_vector_type(8)))  float  v8f;

__device__ __forceinline__ v8f wmma_bf16(v16bf a, v16bf b, v8f c) {
  // 8 args: (neg_a, A, neg_b, B, c_mod, C, reuse_a, reuse_b)
  return __builtin_amdgcn_wmma_f32_16x16x32_bf16(false, a, false, b, (short)0, c,
                                                 false, false);
}

__device__ __forceinline__ v8f v8f_zero() {
  v8f z;
#pragma unroll
  for (int i = 0; i < 8; ++i) z[i] = 0.0f;
  return z;
}

// K index mapping for 16-bit A/B fragments (ISA 7.12.2, 16-bit 16x32):
// dword p (pair of bf16), half = lane>>4:  k = (p>>2)*16 + half*8 + (p&3)*2
__device__ __forceinline__ int wmma_kmap(int p, int half) {
  return ((p >> 2) << 4) | (half << 3) | ((p & 3) << 1);
}

union FragU {
  v16bf bf;
  uint32_t u[8];
};

// A fragment (16x32, MxK), row-major source: elem(m,k)=base[m*ld+k].
__device__ __forceinline__ v16bf fragA_rm(const __bf16* base, int ld, int lane) {
  const int m = lane & 15, half = (lane >> 4) & 1;
  FragU f;
#pragma unroll
  for (int p = 0; p < 8; ++p) {
    const int k = wmma_kmap(p, half);
    f.u[p] = *(const uint32_t*)(base + m * ld + k);
  }
  return f.bf;
}
// B fragment (32x16, KxN) from transposed source: elem(k,n)=base[n*ld+k].
__device__ __forceinline__ v16bf fragB_tr(const __bf16* base, int ld, int lane) {
  const int n = lane & 15, half = (lane >> 4) & 1;
  FragU f;
#pragma unroll
  for (int p = 0; p < 8; ++p) {
    const int k = wmma_kmap(p, half);
    f.u[p] = *(const uint32_t*)(base + n * ld + k);
  }
  return f.bf;
}
// A fragment with transpose: elem(m,k)=base[k*ld+m]  (strided; rare path)
__device__ __forceinline__ v16bf fragA_tr(const __bf16* base, int ld, int lane) {
  const int m = lane & 15, half = (lane >> 4) & 1;
  v16bf f;
#pragma unroll
  for (int p = 0; p < 8; ++p) {
    const int k = wmma_kmap(p, half);
    f[2 * p]     = base[k * ld + m];
    f[2 * p + 1] = base[(k + 1) * ld + m];
  }
  return f;
}

// ---------------------------------------------------------------------------
// fp32 GEMM via bf16 WMMA: C[M,N] = A[M,K] * B[K,N]
// Requirements: M % 128 == 0, K % 32 == 0 (true for all call sites); N ragged.
// 128x128 block tile, K-tile 32, register-staged LDS double buffer (one
// barrier per K-step). LDS tiles in fragment-packed order [subtile][lane][16]
// so a fragment load is one 32B vector (2x ds_load_b128).
// ---------------------------------------------------------------------------
__global__ __launch_bounds__(256) void gemm_f32_bf16wmma(
    const float* __restrict__ A, const float* __restrict__ Bm,
    float* __restrict__ Cm, int M, int N, int K) {
  __shared__ __align__(32) __bf16 sA[2][8 * 32 * 16];  // 2 x 8KB
  __shared__ __align__(32) __bf16 sB[2][8 * 32 * 16];  // 2 x 8KB
  const int tid = threadIdx.x;
  const int lane = tid & 31, wid = tid >> 5;
  const int bm = blockIdx.y * 128, bn = blockIdx.x * 128;
  const int wr = (wid >> 1) * 32;  // wave row offset (4 row-waves)
  const int wc = (wid & 1) * 64;   // wave col offset (2 col-waves)

  // Per-thread staging octets (k-octet = 8 consecutive k of one row/col).
  int amRow[2], akq[2], bnCol[2], bkq[2], bgn[2];
  size_t aBase[2];
  bool bok[2];
#pragma unroll
  for (int oo = 0; oo < 2; ++oo) {
    const int o = tid + oo * 256;
    amRow[oo] = o >> 2;  akq[oo] = o & 3;          // A: m = o/4, k-octet = o%4
    aBase[oo] = (size_t)(bm + amRow[oo]) * K + akq[oo] * 8;
    bnCol[oo] = o & 127; bkq[oo] = o >> 7;         // B: n = o%128, k-octet = o/128
    bgn[oo] = bn + bnCol[oo];
    bok[oo] = bgn[oo] < N;
  }

  float ra[2][8], rb[2][8];
  auto fetch = [&](int k0) {
#pragma unroll
    for (int oo = 0; oo < 2; ++oo) {  // A: coalesced float4 pairs, no guards
      const float4 v0 = *(const float4*)(A + aBase[oo] + k0);
      const float4 v1 = *(const float4*)(A + aBase[oo] + k0 + 4);
      ra[oo][0] = v0.x; ra[oo][1] = v0.y; ra[oo][2] = v0.z; ra[oo][3] = v0.w;
      ra[oo][4] = v1.x; ra[oo][5] = v1.y; ra[oo][6] = v1.z; ra[oo][7] = v1.w;
    }
#pragma unroll
    for (int oo = 0; oo < 2; ++oo) {  // B: per-k-row coalesced, octet-level guard
      if (bok[oo]) {
        const size_t base = (size_t)(k0 + bkq[oo] * 8) * N + bgn[oo];
#pragma unroll
        for (int j = 0; j < 8; ++j) rb[oo][j] = Bm[base + (size_t)j * N];
      } else {
#pragma unroll
        for (int j = 0; j < 8; ++j) rb[oo][j] = 0.0f;
      }
    }
  };
  auto stage = [&](int buf) {
#pragma unroll
    for (int oo = 0; oo < 2; ++oo) {
      bf16x8 pa, pb;
#pragma unroll
      for (int j = 0; j < 8; ++j) { pa[j] = (__bf16)ra[oo][j]; pb[j] = (__bf16)rb[oo][j]; }
      const int sta = amRow[oo] >> 4;
      const int lna = (amRow[oo] & 15) + ((akq[oo] & 1) << 4);
      *(bf16x8*)&sA[buf][(sta * 32 + lna) * 16 + ((akq[oo] >> 1) << 3)] = pa;
      const int stb = bnCol[oo] >> 4;
      const int lnb = (bnCol[oo] & 15) + ((bkq[oo] & 1) << 4);
      *(bf16x8*)&sB[buf][(stb * 32 + lnb) * 16 + ((bkq[oo] >> 1) << 3)] = pb;
    }
  };

  v8f acc[2][4];
#pragma unroll
  for (int mi = 0; mi < 2; ++mi)
#pragma unroll
    for (int ni = 0; ni < 4; ++ni) acc[mi][ni] = v8f_zero();

  const int kt = K >> 5;
  fetch(0);
  stage(0);
  for (int t = 0; t < kt; ++t) {
    __syncthreads();  // staged writes for tile t visible; prev reads done
    const int buf = t & 1;
    if (t + 1 < kt) fetch((t + 1) << 5);  // overlap global loads with WMMA

    v16bf bfrag[4];
#pragma unroll
    for (int ni = 0; ni < 4; ++ni)
      bfrag[ni] = *(const v16bf*)&sB[buf][(((wc >> 4) + ni) * 32 + lane) * 16];
#pragma unroll
    for (int mi = 0; mi < 2; ++mi) {
      const v16bf afrag = *(const v16bf*)&sA[buf][(((wr >> 4) + mi) * 32 + lane) * 16];
#pragma unroll
      for (int ni = 0; ni < 4; ++ni)
        acc[mi][ni] = wmma_bf16(afrag, bfrag[ni], acc[mi][ni]);
    }
    if (t + 1 < kt) stage(buf ^ 1);  // write other buffer; no extra barrier
  }

  // store C (C/D layout: VGPR r -> M=r(+8 for lanes 16..31), N=lane%16)
#pragma unroll
  for (int mi = 0; mi < 2; ++mi) {
#pragma unroll
    for (int ni = 0; ni < 4; ++ni) {
      const int n = bn + wc + ni * 16 + (lane & 15);
      const int mb = bm + wr + mi * 16 + ((lane >> 4) << 3);
      if (n < N) {
#pragma unroll
        for (int r = 0; r < 8; ++r)
          Cm[(size_t)(mb + r) * N + n] = acc[mi][ni][r];
      }
    }
  }
}

// ---------------------------------------------------------------------------
// Elementwise kernels (shapes hardcoded: B=2, S=4096, H*D=2048)
// ---------------------------------------------------------------------------
__global__ __launch_bounds__(128) void l2norm_rows128(float* __restrict__ x) {
  const int row = blockIdx.x, d = threadIdx.x;
  __shared__ float red[128];
  const float v = x[(size_t)row * 128 + d];
  red[d] = v * v;
  __syncthreads();
#pragma unroll
  for (int s = 64; s > 0; s >>= 1) {
    if (d < s) red[d] += red[d + s];
    __syncthreads();
  }
  x[(size_t)row * 128 + d] = v * rsqrtf(red[0] + 1e-6f);
}

__global__ __launch_bounds__(256) void causal_conv_silu(
    const float* __restrict__ x, const float* __restrict__ w,
    float* __restrict__ y) {
  const size_t total = (size_t)2 * 4096 * 2048;
  for (size_t i = (size_t)blockIdx.x * blockDim.x + threadIdx.x; i < total;
       i += (size_t)gridDim.x * blockDim.x) {
    const int ch = (int)(i & 2047);
    const int s = (int)((i >> 11) & 4095);
    float acc = 0.0f;
#pragma unroll
    for (int j = 0; j < 4; ++j) {
      if (s >= 3 - j)  // causal pad
        acc += w[j * 2048 + ch] * x[i - (size_t)(3 - j) * 2048];
    }
    y[i] = acc / (1.0f + __expf(-acc));  // SiLU
  }
}

__global__ __launch_bounds__(256) void sigmoid_inplace(float* __restrict__ x, size_t n) {
  for (size_t i = (size_t)blockIdx.x * blockDim.x + threadIdx.x; i < n;
       i += (size_t)gridDim.x * blockDim.x)
    x[i] = 1.0f / (1.0f + __expf(-x[i]));
}

// g = -exp(A_log[h]) * softplus(glin + dt_bias)   (elementwise, in-place safe)
__global__ __launch_bounds__(256) void kda_gate_kernel(
    const float* __restrict__ glin, const float* __restrict__ A_log,
    const float* __restrict__ dt_bias, float* __restrict__ g, size_t rows) {
  const size_t total = rows * 2048;
  for (size_t i = (size_t)blockIdx.x * blockDim.x + threadIdx.x; i < total;
       i += (size_t)gridDim.x * blockDim.x) {
    const int ch = (int)(i & 2047);
    const int h = ch >> 7;
    const float xv = glin[i] + dt_bias[ch];
    const float sp = (xv > 20.0f) ? xv : log1pf(__expf(xv));
    g[i] = -__expf(A_log[h]) * sp;
  }
}

// gated RMSNorm over head_dim: y = o * rsqrt(mean(o^2)+eps) * scale * sigmoid(gate)
__global__ __launch_bounds__(128) void rms_gate_kernel(
    const float* __restrict__ o, const float* __restrict__ gate,
    const float* __restrict__ scale, float* __restrict__ y) {
  const int row = blockIdx.x, d = threadIdx.x;
  __shared__ float red[128];
  const float v = o[(size_t)row * 128 + d];
  red[d] = v * v;
  __syncthreads();
#pragma unroll
  for (int s = 64; s > 0; s >>= 1) {
    if (d < s) red[d] += red[d + s];
    __syncthreads();
  }
  const float inv = rsqrtf(red[0] * (1.0f / 128.0f) + 1e-5f);
  const float gt = gate[(size_t)row * 128 + d];
  y[(size_t)row * 128 + d] = v * inv * scale[d] * (1.0f / (1.0f + __expf(-gt)));
}

// ---------------------------------------------------------------------------
// Chunked delta-rule scan. One 256-thread block (8 waves) per (b,h).
// Sequential over NC chunks; state S kept transposed in LDS ([dv][d], bf16)
// so all B-role fragments take the contiguous-dword path and C tiles use
// vector b128 loads/stores. All matmuls -> v_wmma_f32_16x16x32_bf16.
// ---------------------------------------------------------------------------
__global__ __launch_bounds__(256) void kda_scan_kernel(
    const float* __restrict__ Q, const float* __restrict__ Km,
    const float* __restrict__ V, const float* __restrict__ G,
    const float* __restrict__ BETA, float* __restrict__ O, int B, int S) {
  __shared__ __align__(32) __bf16 s_kl[64 * 128];   // k*exp(gcs)      [t][d]
  __shared__ __align__(32) __bf16 s_klT[128 * 64];  // same, transposed [d][t]
  __shared__ __align__(32) __bf16 s_kr[64 * 128];   // k*exp(-gcs)     [t][d]
  __shared__ __align__(32) __bf16 s_ql[64 * 128];   // q*scale*exp(gcs)[t][d]
  __shared__ __align__(32) __bf16 s_vT[128 * 64];   // v / v_new       [dv][t]
  __shared__ __align__(32) __bf16 s_wn[64 * 128];   // -w [t][d]; attn aliased
  __shared__ float s_M[64 * 64];                    // -A, then A_inv
  __shared__ __align__(32) __bf16 s_T[64 * 64];     // T               [t][t']
  __shared__ __align__(32) __bf16 s_ST[128 * 128];  // state S^T       [dv][d]
  __shared__ float s_decay[128];
  __shared__ float s_beta[64];
  __bf16* s_attn = s_wn;  // attn [t][t2] bf16 (64*64 fits in s_wn)

  const int tid = threadIdx.x, lane = tid & 31, wid = tid >> 5;
  const int b = blockIdx.x >> 4;  // H = 16
  const int h = blockIdx.x & 15;
  const float qscale = 0.088388347648318447f;  // 128^-0.5

  for (int i = tid; i < 128 * 128; i += 256) s_ST[i] = (__bf16)0.0f;
  __syncthreads();

  const int NC = S / 64;
  for (int c = 0; c < NC; ++c) {
    const int s0 = c * 64;

    // ---- phase 0: per-column cumulative gate; decayed q/k tiles; v; beta ----
    if (tid < 128) {
      const int d = tid;
      float gsum = 0.0f;
      for (int t = 0; t < 64; ++t) {
        const size_t gi = ((size_t)(b * S + s0 + t) * 16 + h) * 128 + d;
        gsum += G[gi];
        const float e = __expf(gsum), en = __expf(-gsum);
        const float kv = Km[gi], qv = Q[gi];
        const __bf16 klv = (__bf16)(kv * e);
        s_kl[t * 128 + d] = klv;
        s_klT[d * 64 + t] = klv;
        s_kr[t * 128 + d] = (__bf16)(kv * en);
        s_ql[t * 128 + d] = (__bf16)(qv * qscale * e);
        if (t == 63) s_decay[d] = e;
      }
    } else if (tid < 192) {
      const int t = tid - 128;
      s_beta[t] = BETA[(size_t)(b * S + s0 + t) * 16 + h];
    }
    for (int i = tid; i < 64 * 128; i += 256) {
      const int t = i >> 7, d = i & 127;
      s_vT[d * 64 + t] = (__bf16)V[((size_t)(b * S + s0 + t) * 16 + h) * 128 + d];
    }
    __syncthreads();

    // ---- A = tril_strict(kl @ kr^T) * beta[row];  store M = -A ----
    {
      const int tr = wid >> 1;
      v8f acc[2] = {v8f_zero(), v8f_zero()};
#pragma unroll
      for (int kk = 0; kk < 4; ++kk) {
        const v16bf a = fragA_rm(s_kl + (tr * 16) * 128 + kk * 32, 128, lane);
#pragma unroll
        for (int j = 0; j < 2; ++j) {
          const int tc = (wid & 1) * 2 + j;
          const v16bf bf = fragB_tr(s_kr + (tc * 16) * 128 + kk * 32, 128, lane);
          acc[j] = wmma_bf16(a, bf, acc[j]);
        }
      }
#pragma unroll
      for (int j = 0; j < 2; ++j) {
        const int tc = (wid & 1) * 2 + j;
        const int n = tc * 16 + (lane & 15);
        const int mb = tr * 16 + ((lane >> 4) << 3);
#pragma unroll
        for (int r = 0; r < 8; ++r) {
          const int m = mb + r;
          s_M[m * 64 + n] = (m > n) ? (-acc[j][r] * s_beta[m]) : 0.0f;
        }
      }
    }
    __syncthreads();

    // ---- serial forward substitution: A_inv in s_M ----
    for (int i = 1; i < 64; ++i) {
      float val = 0.0f;
      const int j = tid;
      const bool act = (tid < 64) && (j < i);
      if (act) {
        float inc = s_M[i * 64 + j];
        for (int l = 0; l < i; ++l) inc += s_M[i * 64 + l] * s_M[l * 64 + j];
        val = inc;
      }
      __syncthreads();
      if (act) s_M[i * 64 + j] = val;
      __syncthreads();
    }
    // T = (A_inv + I) * beta[col]  (bf16)
    for (int i = tid; i < 64 * 64; i += 256) {
      const int ri = i >> 6, cj = i & 63;
      s_T[i] = (__bf16)((s_M[i] + (ri == cj ? 1.0f : 0.0f)) * s_beta[cj]);
    }
    __syncthreads();

    // ---- u = T @ v ; w = T @ kl ; store -w ; v_new = u + (-w) @ S ----
    {
      const int tr = wid & 3;
      const int cg = (wid >> 2) * 64;
      v8f uacc[4], wacc[4];
#pragma unroll
      for (int j = 0; j < 4; ++j) { uacc[j] = v8f_zero(); wacc[j] = v8f_zero(); }
#pragma unroll
      for (int kk = 0; kk < 2; ++kk) {
        const v16bf a = fragA_rm(s_T + (tr * 16) * 64 + kk * 32, 64, lane);
#pragma unroll
        for (int j = 0; j < 4; ++j) {
          const v16bf bv = fragB_tr(s_vT + (cg + j * 16) * 64 + kk * 32, 64, lane);
          uacc[j] = wmma_bf16(a, bv, uacc[j]);
          const v16bf bk = fragB_tr(s_klT + (cg + j * 16) * 64 + kk * 32, 64, lane);
          wacc[j] = wmma_bf16(a, bk, wacc[j]);
        }
      }
#pragma unroll
      for (int j = 0; j < 4; ++j) {
        const int n = cg + j * 16 + (lane & 15);
        const int mb = tr * 16 + ((lane >> 4) << 3);
#pragma unroll
        for (int r = 0; r < 8; ++r) s_wn[(mb + r) * 128 + n] = (__bf16)(-wacc[j][r]);
      }
      __syncthreads();
#pragma unroll
      for (int kk = 0; kk < 4; ++kk) {
        const v16bf a = fragA_rm(s_wn + (tr * 16) * 128 + kk * 32, 128, lane);
#pragma unroll
        for (int j = 0; j < 4; ++j) {
          const v16bf bs = fragB_tr(s_ST + (cg + j * 16) * 128 + kk * 32, 128, lane);
          uacc[j] = wmma_bf16(a, bs, uacc[j]);
        }
      }
      // v_new -> s_vT (vector b128 stores; all u-phase reads of s_vT done
      // before the barrier above)
#pragma unroll
      for (int j = 0; j < 4; ++j) {
        const int n = cg + j * 16 + (lane & 15);
        const int mb = tr * 16 + ((lane >> 4) << 3);
        bf16x8 pk;
#pragma unroll
        for (int r = 0; r < 8; ++r) pk[r] = (__bf16)uacc[j][r];
        *(bf16x8*)&s_vT[n * 64 + mb] = pk;
      }
    }
    __syncthreads();

    // ---- attn = tril(ql @ kr^T)  (bf16, aliased over s_wn) ----
    {
      const int tr = wid >> 1;
      v8f acc[2] = {v8f_zero(), v8f_zero()};
#pragma unroll
      for (int kk = 0; kk < 4; ++kk) {
        const v16bf a = fragA_rm(s_ql + (tr * 16) * 128 + kk * 32, 128, lane);
#pragma unroll
        for (int j = 0; j < 2; ++j) {
          const int tc = (wid & 1) * 2 + j;
          const v16bf bf = fragB_tr(s_kr + (tc * 16) * 128 + kk * 32, 128, lane);
          acc[j] = wmma_bf16(a, bf, acc[j]);
        }
      }
      // s_wn reads (v_new phase) completed before previous barrier
#pragma unroll
      for (int j = 0; j < 2; ++j) {
        const int tc = (wid & 1) * 2 + j;
        const int n = tc * 16 + (lane & 15);
        const int mb = tr * 16 + ((lane >> 4) << 3);
#pragma unroll
        for (int r = 0; r < 8; ++r) {
          const int m = mb + r;
          s_attn[m * 64 + n] = (__bf16)((m >= n) ? acc[j][r] : 0.0f);
        }
      }
    }
    __syncthreads();

    // ---- o = ql @ S + attn @ v_new ; write to global ----
    {
      const int tr = wid & 3;
      const int cg = (wid >> 2) * 64;
      v8f oacc[4];
#pragma unroll
      for (int j = 0; j < 4; ++j) oacc[j] = v8f_zero();
#pragma unroll
      for (int kk = 0; kk < 4; ++kk) {  // ql @ S (K = 128)
        const v16bf a = fragA_rm(s_ql + (tr * 16) * 128 + kk * 32, 128, lane);
#pragma unroll
        for (int j = 0; j < 4; ++j) {
          const v16bf bs = fragB_tr(s_ST + (cg + j * 16) * 128 + kk * 32, 128, lane);
          oacc[j] = wmma_bf16(a, bs, oacc[j]);
        }
      }
#pragma unroll
      for (int kk = 0; kk < 2; ++kk) {  // attn @ v_new (K = 64)
        const v16bf a = fragA_rm(s_attn + (tr * 16) * 64 + kk * 32, 64, lane);
#pragma unroll
        for (int j = 0; j < 4; ++j) {
          const v16bf bv = fragB_tr(s_vT + (cg + j * 16) * 64 + kk * 32, 64, lane);
          oacc[j] = wmma_bf16(a, bv, oacc[j]);
        }
      }
#pragma unroll
      for (int j = 0; j < 4; ++j) {
        const int n = cg + j * 16 + (lane & 15);
        const int mb = tr * 16 + ((lane >> 4) << 3);
#pragma unroll
        for (int r = 0; r < 8; ++r) {
          const int t = s0 + mb + r;
          O[((size_t)(b * S + t) * 16 + h) * 128 + n] = oacc[j][r];
        }
      }
    }
    __syncthreads();

    // ---- S^T = decay[d] * (S^T + (kr^T @ v_new)^T), tiles in C layout ----
    {
      const int tr = wid;  // 8 row tiles of d
      v8f sacc[8];
#pragma unroll
      for (int j = 0; j < 8; ++j) {  // load C tiles (vector b128 from S^T)
        const int n = j * 16 + (lane & 15);
        const int mb = tr * 16 + ((lane >> 4) << 3);
        const bf16x8 raw = *(const bf16x8*)&s_ST[n * 128 + mb];
        v8f cvals;
#pragma unroll
        for (int r = 0; r < 8; ++r) cvals[r] = (float)raw[r];
        sacc[j] = cvals;
      }
#pragma unroll
      for (int kk = 0; kk < 2; ++kk) {
        const v16bf a = fragA_tr(s_kr + (kk * 32) * 128 + tr * 16, 128, lane);
#pragma unroll
        for (int j = 0; j < 8; ++j) {
          const v16bf bv = fragB_tr(s_vT + (j * 16) * 64 + kk * 32, 64, lane);
          sacc[j] = wmma_bf16(a, bv, sacc[j]);
        }
      }
#pragma unroll
      for (int j = 0; j < 8; ++j) {
        const int n = j * 16 + (lane & 15);
        const int mb = tr * 16 + ((lane >> 4) << 3);
        bf16x8 pk;
#pragma unroll
        for (int r = 0; r < 8; ++r)
          pk[r] = (__bf16)(s_decay[mb + r] * sacc[j][r]);
        *(bf16x8*)&s_ST[n * 128 + mb] = pk;
      }
    }
    __syncthreads();
  }
}

// ---------------------------------------------------------------------------
// Host-side launcher
// ---------------------------------------------------------------------------
static inline void launch_gemm(const float* A, const float* B, float* C, int M,
                               int N, int K, hipStream_t st) {
  dim3 grid((N + 127) / 128, (M + 127) / 128);
  gemm_f32_bf16wmma<<<grid, 256, 0, st>>>(A, B, C, M, N, K);
}

extern "C" void kernel_launch(void* const* d_in, const int* in_sizes, int n_in,
                              void* d_out, int out_size, void* d_ws, size_t ws_size,
                              hipStream_t stream) {
  (void)in_sizes; (void)n_in; (void)out_size; (void)ws_size;
  const float* hs       = (const float*)d_in[0];
  const float* Wq       = (const float*)d_in[1];
  const float* Wk       = (const float*)d_in[2];
  const float* Wv       = (const float*)d_in[3];
  const float* conv_q   = (const float*)d_in[4];
  const float* conv_k   = (const float*)d_in[5];
  const float* conv_v   = (const float*)d_in[6];
  const float* Wb       = (const float*)d_in[7];
  const float* Wfa      = (const float*)d_in[8];
  const float* Wfb      = (const float*)d_in[9];
  const float* Wga      = (const float*)d_in[10];
  const float* Wgb      = (const float*)d_in[11];
  const float* A_log    = (const float*)d_in[12];
  const float* dt_bias  = (const float*)d_in[13];
  const float* rms_scale= (const float*)d_in[14];
  const float* Wo       = (const float*)d_in[15];
  // d_in[16] = chunk_size (64; hardwired)

  const int B = 2, S = 4096, HS = 2048, H = 16;
  const size_t BS  = (size_t)B * S;        // 8192
  const size_t BIG = BS * 2048;            // elements per [B,S,H*D] tensor

  float* ws   = (float*)d_ws;
  float* buf0 = ws + 0 * BIG;  // q proj -> later g_lin/g
  float* buf1 = ws + 1 * BIG;  // k proj -> later o
  float* buf2 = ws + 2 * BIG;  // v proj -> later gate
  float* buf3 = ws + 3 * BIG;  // q (conv+silu) -> later y
  float* buf4 = ws + 4 * BIG;  // k (conv+silu)
  float* buf5 = ws + 5 * BIG;  // v (conv+silu)
  float* beta = ws + 6 * BIG;              // BS * 16
  float* fa   = beta + BS * 16;            // BS * 128
  float* ga   = fa + BS * 128;             // BS * 128

  // q/k/v projections + l2norm + causal conv + SiLU
  launch_gemm(hs, Wq, buf0, (int)BS, 2048, HS, stream);
  l2norm_rows128<<<(int)(BS * 16), 128, 0, stream>>>(buf0);
  causal_conv_silu<<<4096, 256, 0, stream>>>(buf0, conv_q, buf3);

  launch_gemm(hs, Wk, buf1, (int)BS, 2048, HS, stream);
  l2norm_rows128<<<(int)(BS * 16), 128, 0, stream>>>(buf1);
  causal_conv_silu<<<4096, 256, 0, stream>>>(buf1, conv_k, buf4);

  launch_gemm(hs, Wv, buf2, (int)BS, 2048, HS, stream);
  causal_conv_silu<<<4096, 256, 0, stream>>>(buf2, conv_v, buf5);

  // beta = sigmoid(hs @ Wb)
  launch_gemm(hs, Wb, beta, (int)BS, 16, HS, stream);
  sigmoid_inplace<<<512, 256, 0, stream>>>(beta, BS * 16);

  // g = -exp(A_log) * softplus(hs @ Wfa @ Wfb + dt_bias)
  launch_gemm(hs, Wfa, fa, (int)BS, 128, HS, stream);
  launch_gemm(fa, Wfb, buf0, (int)BS, 2048, 128, stream);
  kda_gate_kernel<<<8192, 256, 0, stream>>>(buf0, A_log, dt_bias, buf0, BS);

  // chunked delta-rule scan -> o in buf1
  kda_scan_kernel<<<B * H, 256, 0, stream>>>(buf3, buf4, buf5, buf0, beta, buf1, B, S);

  // gate = hs @ Wga @ Wgb
  launch_gemm(hs, Wga, ga, (int)BS, 128, HS, stream);
  launch_gemm(ga, Wgb, buf2, (int)BS, 2048, 128, stream);

  // y = RMSNormGated(o, gate) -> buf3 ; out = y @ Wo
  rms_gate_kernel<<<(int)(BS * 16), 128, 0, stream>>>(buf1, buf2, rms_scale, buf3);
  launch_gemm(buf3, Wo, (float*)d_out, (int)BS, HS, 2048, stream);
}
